// RandlaNet_34849364640184
// MI455X (gfx1250) — matrix operand order repaired and checked by reading, not compiled, not run on hardware
//
#include <hip/hip_runtime.h>

typedef __attribute__((ext_vector_type(16))) _Float16 v16h;
typedef __attribute__((ext_vector_type(8)))  float    v8f;

#define N_PTS 200000
#define KNN   16

#if __has_builtin(__builtin_amdgcn_global_load_async_to_lds_b128)
#define HAS_ASYNC_LDS 1
typedef int int4v __attribute__((ext_vector_type(4)));
typedef __attribute__((address_space(1))) int4v g_i4;   // global (device) int4*
typedef __attribute__((address_space(3))) int4v l_i4;   // LDS int4*
#else
#define HAS_ASYNC_LDS 0
#endif

union Frag16 { v16h v; _Float16 h[16]; unsigned int u[8]; uint4 q[2]; };

__device__ __forceinline__ float leaky(float x) { return fmaxf(x, 0.2f * x); }

__device__ __forceinline__ void wait_async0() {
#if __has_builtin(__builtin_amdgcn_s_wait_asynccnt)
    __builtin_amdgcn_s_wait_asynccnt(0);
#else
    asm volatile("s_wait_asynccnt 0x0" ::: "memory");
#endif
}

// ---------------- Kernel 1: h0 = leaky(x @ W_start + b_start), f16 out ----------------
__global__ __launch_bounds__(256) void k_start(const float* __restrict__ x,
                                               const float* __restrict__ Wst,
                                               const float* __restrict__ bst,
                                               _Float16* __restrict__ h0) {
    int lane = threadIdx.x & 31, wave = threadIdx.x >> 5;
    int hh = lane >> 4, n = lane & 15, m = n;
    long i0 = ((long)blockIdx.x * 8 + wave) * 16;
    if (i0 >= N_PTS) return;

    // B: W_start [16x16] zero-padded to 32x16 (clamped-index loads + select: no branches)
    Frag16 b;
#pragma unroll
    for (int v = 0; v < 8; ++v) {
        int k = (v < 4) ? 8 * hh + 2 * v : 16 + 8 * hh + 2 * (v - 4);
        float w0 = Wst[(k < 16 ? k : 15) * 16 + n];
        float w1 = Wst[(k + 1 < 16 ? k + 1 : 15) * 16 + n];
        b.h[2 * v]     = (_Float16)(k     < 16 ? w0 : 0.0f);
        b.h[2 * v + 1] = (_Float16)(k + 1 < 16 ? w1 : 0.0f);
    }
    // A: 16 points x 16 features, K zero-padded to 32; 2x float4 loads per lane
    Frag16 a;
    const float4* xr = (const float4*)(x + (i0 + m) * 16 + 8 * hh);
    float4 x0 = xr[0], x1 = xr[1];
    a.h[0] = (_Float16)x0.x; a.h[1] = (_Float16)x0.y;
    a.h[2] = (_Float16)x0.z; a.h[3] = (_Float16)x0.w;
    a.h[4] = (_Float16)x1.x; a.h[5] = (_Float16)x1.y;
    a.h[6] = (_Float16)x1.z; a.h[7] = (_Float16)x1.w;
    a.q[1] = make_uint4(0u, 0u, 0u, 0u);

    v8f c = {};
    c = __builtin_amdgcn_wmma_f32_16x16x32_f16(false, a.v, false, b.v, (short)0, c, false, false);
    float bias = bst[n];
#pragma unroll
    for (int v = 0; v < 8; ++v) {
        float val = leaky(c[v] + bias);
        h0[(i0 + v + 8 * hh) * 16 + n] = (_Float16)val;
    }
}

// ---------------- Kernel 2/3: RandLA attention conv, one point per wave ----------------
template <int DOUT>
__global__ __launch_bounds__(256) void k_conv(const _Float16* __restrict__ hin,  // [N,16] f16
                                              const float* __restrict__ pos,     // [N,3]
                                              const int* __restrict__ nidx,      // [N,16]
                                              const float* __restrict__ Wr,      // [10,16]
                                              const float* __restrict__ br,      // [16]
                                              const float* __restrict__ Wa,      // [32,32]
                                              const float* __restrict__ Wp,      // [32,DOUT]
                                              const float* __restrict__ bp,      // [DOUT]
                                              _Float16* __restrict__ hout) {     // [N,DOUT] f16
    __shared__ _Float16 rp_s[8 * 512];   // per-wave 16x32 f16 (K-padded rel-pos)
    __shared__ _Float16 fh_s[8 * 512];   // per-wave 16x32 f16 (f_hat)
    int lane = threadIdx.x & 31, wave = threadIdx.x >> 5;
    int hh = lane >> 4, n = lane & 15, m = n;
    long i = (long)blockIdx.x * 8 + wave;  // grid == N/8 exactly; every wave live

    _Float16* rp_w = rp_s + wave * 512;
    _Float16* fh_w = fh_s + wave * 512;
    unsigned int* rpu = (unsigned int*)rp_w;
    unsigned int* fhu = (unsigned int*)fh_w;

    // zero padding cols of rp (u32 cols 5..15 of each row) — written once
    for (int t = lane; t < 16 * 11; t += 32) {
        int r = t / 11, c2 = 5 + t % 11;
        rpu[r * 16 + c2] = 0u;
    }

    // weight fragments in registers (clamped-index loads, selects instead of branches)
    Frag16 brp, ba0, ba1;
#pragma unroll
    for (int v = 0; v < 8; ++v) {
        int k = (v < 4) ? 8 * hh + 2 * v : 16 + 8 * hh + 2 * (v - 4);
        float wr0 = Wr[(k < 10 ? k : 9) * 16 + n];
        float wr1 = Wr[(k + 1 < 10 ? k + 1 : 9) * 16 + n];
        brp.h[2 * v]     = (_Float16)(k     < 10 ? wr0 : 0.0f);
        brp.h[2 * v + 1] = (_Float16)(k + 1 < 10 ? wr1 : 0.0f);
        ba0.h[2 * v]     = (_Float16)Wa[k * 32 + n];
        ba0.h[2 * v + 1] = (_Float16)Wa[(k + 1) * 32 + n];
        ba1.h[2 * v]     = (_Float16)Wa[k * 32 + 16 + n];
        ba1.h[2 * v + 1] = (_Float16)Wa[(k + 1) * 32 + 16 + n];
    }
    int cc = lane % DOUT;
    float wpost[32];
#pragma unroll
    for (int f = 0; f < 32; ++f) wpost[f] = Wp[f * DOUT + cc];
    float bpost = bp[cc];
    float brb = br[n];

    // phase 1: gather neighbor features (async global->LDS) + rel-pos encoding inputs
    int kn = n;                                  // neighbor index handled by this lane
    int j = nidx[i * KNN + kn];
    {
        const _Float16* gsrc = hin + (long)j * 16 + 8 * hh;
        _Float16* ldst = fh_w + kn * 32 + 8 * hh;
#if HAS_ASYNC_LDS
        unsigned long long ga = (unsigned long long)(const void*)gsrc;
        unsigned la = (unsigned)(unsigned long long)(const void*)ldst;  // LDS_ADDR = flat[31:0]
        __builtin_amdgcn_global_load_async_to_lds_b128((g_i4*)ga, (l_i4*)la, 0, 0);
#else
        *((uint4*)ldst) = *((const uint4*)gsrc);
#endif
    }
    if (lane < 16) {
        float pix = pos[i * 3 + 0], piy = pos[i * 3 + 1], piz = pos[i * 3 + 2];
        float pjx = pos[(long)j * 3 + 0], pjy = pos[(long)j * 3 + 1], pjz = pos[(long)j * 3 + 2];
        float dx = pix - pjx, dy = piy - pjy, dz = piz - pjz;
        float dist = sqrtf(dx * dx + dy * dy + dz * dz + 1e-12f);
        union { uint4 q; _Float16 hp[8]; } pk;
        pk.hp[0] = (_Float16)pix; pk.hp[1] = (_Float16)piy;
        pk.hp[2] = (_Float16)piz; pk.hp[3] = (_Float16)pjx;
        pk.hp[4] = (_Float16)pjy; pk.hp[5] = (_Float16)pjz;
        pk.hp[6] = (_Float16)dx;  pk.hp[7] = (_Float16)dy;
        *((uint4*)(rpu + kn * 16)) = pk.q;                 // cols 0..7 (b128)
        union { unsigned int u; _Float16 hp[2]; } pk2;
        pk2.hp[0] = (_Float16)dz; pk2.hp[1] = (_Float16)dist;
        rpu[kn * 16 + 4] = pk2.u;                          // cols 8..9 (b32)
    }
#if HAS_ASYNC_LDS
    wait_async0();
#endif
    __syncthreads();

    // phase 2: r = leaky(rp @ Wr + br) via WMMA; write into f_hat cols 16..31
    Frag16 a1;   // two contiguous 16B runs per lane
    a1.q[0] = *((const uint4*)(rpu + m * 16 + 4 * hh));
    a1.q[1] = *((const uint4*)(rpu + m * 16 + 8 + 4 * hh));
    v8f zero8 = {};
    v8f d1 = __builtin_amdgcn_wmma_f32_16x16x32_f16(false, a1.v, false, brp.v, (short)0, zero8, false, false);
    float rkeep[8];
#pragma unroll
    for (int v = 0; v < 8; ++v) {
        float r = leaky(d1[v] + brb);
        _Float16 rh = (_Float16)r;
        rkeep[v] = (float)rh;                         // f16-rounded, same as WMMA input
        fh_w[(v + 8 * hh) * 32 + 16 + n] = rh;
    }
    __syncthreads();

    // phase 3: attention logits = f_hat @ W_att (two 16-col tiles)
    Frag16 a2;
    a2.q[0] = *((const uint4*)(fhu + m * 16 + 4 * hh));
    a2.q[1] = *((const uint4*)(fhu + m * 16 + 8 + 4 * hh));
    v8f da = __builtin_amdgcn_wmma_f32_16x16x32_f16(false, a2.v, false, ba0.v, (short)0, zero8, false, false);
    v8f db = __builtin_amdgcn_wmma_f32_16x16x32_f16(false, a2.v, false, ba1.v, (short)0, zero8, false, false);

    // phase 4: softmax over neighbor (row) axis + weighted aggregation
    float aggA, aggB;
    {
        float mx = -3.0e38f;
#pragma unroll
        for (int v = 0; v < 8; ++v) mx = fmaxf(mx, da[v]);
        mx = fmaxf(mx, __shfl_xor(mx, 16, 32));
        float e[8], s = 0.f;
#pragma unroll
        for (int v = 0; v < 8; ++v) { e[v] = __expf(da[v] - mx); s += e[v]; }
        s += __shfl_xor(s, 16, 32);
        float part = 0.f;
#pragma unroll
        for (int v = 0; v < 8; ++v) part += e[v] * (float)fh_w[(v + 8 * hh) * 32 + n];
        part += __shfl_xor(part, 16, 32);
        aggA = part / s;
    }
    {
        float mx = -3.0e38f;
#pragma unroll
        for (int v = 0; v < 8; ++v) mx = fmaxf(mx, db[v]);
        mx = fmaxf(mx, __shfl_xor(mx, 16, 32));
        float e[8], s = 0.f;
#pragma unroll
        for (int v = 0; v < 8; ++v) { e[v] = __expf(db[v] - mx); s += e[v]; }
        s += __shfl_xor(s, 16, 32);
        float part = 0.f;
#pragma unroll
        for (int v = 0; v < 8; ++v) part += e[v] * rkeep[v];   // own phase-2 values: no LDS reads
        part += __shfl_xor(part, 16, 32);
        aggB = part / s;
    }

    // phase 5: hout = leaky(agg[1x32] @ Wp + bp)
    float acc = bpost;
#pragma unroll
    for (int f = 0; f < 16; ++f) acc += __shfl(aggA, f, 32) * wpost[f];
#pragma unroll
    for (int f = 0; f < 16; ++f) acc += __shfl(aggB, f, 32) * wpost[16 + f];
    acc = leaky(acc);
    if (lane < DOUT) hout[i * DOUT + lane] = (_Float16)acc;
}

// ---------------- Kernel 4: out = leaky(h2@W_end + b_end + x@W_skip + b_skip) ----------------
__global__ __launch_bounds__(256) void k_final(const _Float16* __restrict__ h2,  // [N,32] f16
                                               const float* __restrict__ x,      // [N,16]
                                               const float* __restrict__ We,     // [32,64]
                                               const float* __restrict__ be,     // [64]
                                               const float* __restrict__ Wsk,    // [16,64]
                                               const float* __restrict__ bsk,    // [64]
                                               float* __restrict__ out) {        // [N,64]
    int lane = threadIdx.x & 31, wave = threadIdx.x >> 5;
    int hh = lane >> 4, n = lane & 15, m = n;
    long i0 = ((long)blockIdx.x * 8 + wave) * 16;
    if (i0 >= N_PTS) return;

    Frag16 ah;  // h2 tile [16 x 32] f16: two contiguous b128 loads
    const unsigned int* h2u = (const unsigned int*)(h2 + (i0 + m) * 32);
    ah.q[0] = *((const uint4*)(h2u + 4 * hh));
    ah.q[1] = *((const uint4*)(h2u + 8 + 4 * hh));

    Frag16 ax;  // x tile [16 x 16] padded to K=32
    const float4* xr = (const float4*)(x + (i0 + m) * 16 + 8 * hh);
    float4 x0 = xr[0], x1 = xr[1];
    ax.h[0] = (_Float16)x0.x; ax.h[1] = (_Float16)x0.y;
    ax.h[2] = (_Float16)x0.z; ax.h[3] = (_Float16)x0.w;
    ax.h[4] = (_Float16)x1.x; ax.h[5] = (_Float16)x1.y;
    ax.h[6] = (_Float16)x1.z; ax.h[7] = (_Float16)x1.w;
    ax.q[1] = make_uint4(0u, 0u, 0u, 0u);

#pragma unroll
    for (int t = 0; t < 4; ++t) {
        Frag16 bend, bskf;
#pragma unroll
        for (int v = 0; v < 8; ++v) {
            int k = (v < 4) ? 8 * hh + 2 * v : 16 + 8 * hh + 2 * (v - 4);
            int c = 16 * t + n;
            bend.h[2 * v]     = (_Float16)We[k * 64 + c];
            bend.h[2 * v + 1] = (_Float16)We[(k + 1) * 64 + c];
            float ws0 = Wsk[(k < 16 ? k : 15) * 64 + c];
            float ws1 = Wsk[(k + 1 < 16 ? k + 1 : 15) * 64 + c];
            bskf.h[2 * v]     = (_Float16)(k     < 16 ? ws0 : 0.0f);
            bskf.h[2 * v + 1] = (_Float16)(k + 1 < 16 ? ws1 : 0.0f);
        }
        v8f c8 = {};
        // skip path accumulated into C, then chained into the main projection
        c8 = __builtin_amdgcn_wmma_f32_16x16x32_f16(false, ax.v, false, bskf.v, (short)0, c8, false, false);
        c8 = __builtin_amdgcn_wmma_f32_16x16x32_f16(false, ah.v, false, bend.v, (short)0, c8, false, false);
        float bias = be[16 * t + n] + bsk[16 * t + n];
#pragma unroll
        for (int v = 0; v < 8; ++v) {
            float val = leaky(c8[v] + bias);
            out[(i0 + v + 8 * hh) * 64 + 16 * t + n] = val;
        }
    }
}

extern "C" void kernel_launch(void* const* d_in, const int* in_sizes, int n_in,
                              void* d_out, int out_size, void* d_ws, size_t ws_size,
                              hipStream_t stream) {
    const float* x   = (const float*)d_in[0];
    const float* pos = (const float*)d_in[1];
    const int*  nidx = (const int*)d_in[2];
    const float* Wst = (const float*)d_in[3];
    const float* bst = (const float*)d_in[4];
    const float* Wsk = (const float*)d_in[5];
    const float* bsk = (const float*)d_in[6];
    const float* We  = (const float*)d_in[7];
    const float* be  = (const float*)d_in[8];
    const float* Wr1 = (const float*)d_in[9];
    const float* br1 = (const float*)d_in[10];
    const float* Wa1 = (const float*)d_in[11];
    const float* Wp1 = (const float*)d_in[12];
    const float* bp1 = (const float*)d_in[13];
    const float* Wr2 = (const float*)d_in[14];
    const float* br2 = (const float*)d_in[15];
    const float* Wa2 = (const float*)d_in[16];
    const float* Wp2 = (const float*)d_in[17];
    const float* bp2 = (const float*)d_in[18];
    float* out = (float*)d_out;

    char* ws = (char*)d_ws;
    _Float16* h0 = (_Float16*)(ws);                                 // N*16 f16
    _Float16* h1 = (_Float16*)(ws + (size_t)N_PTS * 16 * 2);        // N*16 f16
    _Float16* h2 = (_Float16*)(ws + (size_t)N_PTS * 16 * 4);        // N*32 f16

    dim3 blk(256);
    int g16 = (N_PTS + 127) / 128;  // 16 pts/wave * 8 waves
    int g1  = N_PTS / 8;            // 1 pt/wave * 8 waves (exact: 25000 blocks)

    hipLaunchKernelGGL(k_start, dim3(g16), blk, 0, stream, x, Wst, bst, h0);
    hipLaunchKernelGGL((k_conv<16>), dim3(g1), blk, 0, stream, h0, pos, nidx, Wr1, br1, Wa1, Wp1, bp1, h1);
    hipLaunchKernelGGL((k_conv<32>), dim3(g1), blk, 0, stream, h1, pos, nidx, Wr2, br2, Wa2, Wp2, bp2, h2);
    hipLaunchKernelGGL(k_final, dim3(g16), blk, 0, stream, h2, x, We, be, Wsk, bsk, out);
}
